// Process_x_197568495703
// MI455X (gfx1250) — matrix operand order repaired and checked by reading, not compiled, not run on hardware
//
#include <hip/hip_runtime.h>
#include <math.h>

// ---------------- problem constants ----------------
#define DIMN     1024
#define HEADS    16
#define WINW     128
#define BATCH    4
#define SEQL     4096
#define HEAD_DIM 64
#define HIDDEN   2048
#define ROWS     (BATCH*SEQL)   // 16384

// ---------------- WMMA types ----------------
typedef __attribute__((ext_vector_type(16))) __bf16 bf16x16;
typedef __attribute__((ext_vector_type(8)))  float  floatx8;

struct alignas(16) U4 { unsigned int x, y, z, w; };
union Frag  { bf16x16 v; U4 q[2]; };
union Pack8 { U4 q; __bf16 h[8]; };

__device__ __forceinline__ floatx8 wmma_bf16(const Frag& a, const Frag& b, floatx8 c) {
    return __builtin_amdgcn_wmma_f32_16x16x32_bf16(false, a.v, false, b.v,
                                                   (short)0, c, false, false);
}

// ---------------- fp32 -> bf16 transpose (LDS-tiled, both streams coalesced) ----
// in: [R, C] fp32 row-major.  out: [C, R] bf16 row-major (out[c*R+r] = in[r*C+c]).
__global__ void __launch_bounds__(256)
transpose_bf16_kernel(const float* __restrict__ in, __bf16* __restrict__ out,
                      int R, int C) {
    __shared__ __bf16 t[64][72];   // 72-elem pitch: kills bank conflicts on col reads
    const int tid = threadIdx.x;
    const int c0 = blockIdx.x * 64, r0 = blockIdx.y * 64;
#pragma unroll
    for (int p = 0; p < 4; ++p) {
        int r = (tid >> 4) + 16 * p;
        int c = (tid & 15) * 4;
        const float* src = in + (size_t)(r0 + r) * C + c0 + c;
        t[r][c + 0] = (__bf16)src[0];
        t[r][c + 1] = (__bf16)src[1];
        t[r][c + 2] = (__bf16)src[2];
        t[r][c + 3] = (__bf16)src[3];
    }
    __syncthreads();
#pragma unroll
    for (int it = 0; it < 2; ++it) {
        int s = tid + it * 256;
        int rn = s >> 3, k8 = (s & 7) * 8;
        Pack8 d;
#pragma unroll
        for (int e = 0; e < 8; ++e) d.h[e] = t[k8 + e][rn];
        *reinterpret_cast<U4*>(out + (size_t)(c0 + rn) * R + r0 + k8) = d.q;
    }
}

// ---------------- LayerNorm (one row per block, 256 thr) ----------------
__global__ void __launch_bounds__(256)
layernorm_kernel(const float* __restrict__ X, const float* __restrict__ g,
                 const float* __restrict__ b, __bf16* __restrict__ Y) {
    __shared__ float red[16];
    const int row = blockIdx.x, tid = threadIdx.x;
    const float* xr = X + (size_t)row * DIMN + tid * 4;
    float x0 = xr[0], x1 = xr[1], x2 = xr[2], x3 = xr[3];
    float s  = x0 + x1 + x2 + x3;
    float ss = x0*x0 + x1*x1 + x2*x2 + x3*x3;
    for (int m = 1; m < 32; m <<= 1) { s += __shfl_xor(s, m, 32); ss += __shfl_xor(ss, m, 32); }
    const int wid = tid >> 5, lane = tid & 31;
    if (lane == 0) { red[wid] = s; red[8 + wid] = ss; }
    __syncthreads();
    float ts = 0.f, tss = 0.f;
    for (int w = 0; w < 8; ++w) { ts += red[w]; tss += red[8 + w]; }
    const float mean = ts * (1.0f / DIMN);
    const float var  = tss * (1.0f / DIMN) - mean * mean;
    const float rstd = rsqrtf(var + 1e-5f);
    const int c = tid * 4;
    Y[(size_t)row * DIMN + c + 0] = (__bf16)((x0 - mean) * rstd * g[c + 0] + b[c + 0]);
    Y[(size_t)row * DIMN + c + 1] = (__bf16)((x1 - mean) * rstd * g[c + 1] + b[c + 1]);
    Y[(size_t)row * DIMN + c + 2] = (__bf16)((x2 - mean) * rstd * g[c + 2] + b[c + 2]);
    Y[(size_t)row * DIMN + c + 3] = (__bf16)((x3 - mean) * rstd * g[c + 3] + b[c + 3]);
}

// ---------------- bf16 WMMA GEMM, 128x128 tile, double-buffered LDS ----------
// A: [M,K] bf16 row-major.  Bt: [N,K] bf16 row-major (pre-transposed weights).
// EPI 0: store bf16      EPI 1: f32 = acc + bias[col] + res[row,col]
// EPI 2: f32 out[row,col] += acc + bias[col]
template<int EPI>
__global__ void __launch_bounds__(256)
gemm_wmma(const __bf16* __restrict__ A, const __bf16* __restrict__ Bt,
          int M, int N, int K,
          const float* __restrict__ bias, const float* __restrict__ res,
          __bf16* __restrict__ Cb, float* __restrict__ Cf) {
    __shared__ __bf16 As[2][128 * 32];
    __shared__ __bf16 Bs[2][128 * 32];   // [n][k]
    const int tid = threadIdx.x, lane = tid & 31, wid = tid >> 5;
    const int hi = lane >> 4, ln = lane & 15;
    const int wm = (wid >> 2) * 64, wn = (wid & 3) * 32;
    const int m0 = blockIdx.y * 128, n0 = blockIdx.x * 128;

    floatx8 zero8 = {0.f,0.f,0.f,0.f,0.f,0.f,0.f,0.f};
    floatx8 acc[4][2];
    for (int i = 0; i < 4; ++i) for (int j = 0; j < 2; ++j) acc[i][j] = zero8;

    const int r0 = tid >> 2, c8 = (tid & 3) * 8;          // segment 0
    const int r1 = (tid + 256) >> 2;                       // segment 1 (same c8)
    const __bf16* Ap0 = A  + (size_t)(m0 + r0) * K + c8;
    const __bf16* Ap1 = A  + (size_t)(m0 + r1) * K + c8;
    const __bf16* Bp0 = Bt + (size_t)(n0 + r0) * K + c8;
    const __bf16* Bp1 = Bt + (size_t)(n0 + r1) * K + c8;
    U4 aR[2], bR[2];
    aR[0] = *reinterpret_cast<const U4*>(Ap0);
    aR[1] = *reinterpret_cast<const U4*>(Ap1);
    bR[0] = *reinterpret_cast<const U4*>(Bp0);
    bR[1] = *reinterpret_cast<const U4*>(Bp1);

    const int T = K / 32;
    for (int t = 0; t < T; ++t) {
        const int cur = t & 1;
        *reinterpret_cast<U4*>(&As[cur][r0 * 32 + c8]) = aR[0];
        *reinterpret_cast<U4*>(&As[cur][r1 * 32 + c8]) = aR[1];
        *reinterpret_cast<U4*>(&Bs[cur][r0 * 32 + c8]) = bR[0];
        *reinterpret_cast<U4*>(&Bs[cur][r1 * 32 + c8]) = bR[1];
        if (t + 1 < T) {
            const int kk = (t + 1) * 32;
            aR[0] = *reinterpret_cast<const U4*>(Ap0 + kk);
            aR[1] = *reinterpret_cast<const U4*>(Ap1 + kk);
            bR[0] = *reinterpret_cast<const U4*>(Bp0 + kk);
            bR[1] = *reinterpret_cast<const U4*>(Bp1 + kk);
        }
        if (t + 2 < T) {   // pull t+2 toward the caches (global_prefetch_b8)
            const int kp = (t + 2) * 32;
            __builtin_prefetch(Ap0 + kp, 0, 1);
            __builtin_prefetch(Bp0 + kp, 0, 1);
        }
        __syncthreads();
        Frag aF[4], bF[2];
#pragma unroll
        for (int mt = 0; mt < 4; ++mt) {
            int r = wm + mt * 16 + ln;
            aF[mt].q[0] = *reinterpret_cast<const U4*>(&As[cur][r * 32 + 8 * hi]);
            aF[mt].q[1] = *reinterpret_cast<const U4*>(&As[cur][r * 32 + 16 + 8 * hi]);
        }
#pragma unroll
        for (int nt = 0; nt < 2; ++nt) {
            int c = wn + nt * 16 + ln;
            bF[nt].q[0] = *reinterpret_cast<const U4*>(&Bs[cur][c * 32 + 8 * hi]);
            bF[nt].q[1] = *reinterpret_cast<const U4*>(&Bs[cur][c * 32 + 16 + 8 * hi]);
        }
#pragma unroll
        for (int mt = 0; mt < 4; ++mt)
#pragma unroll
            for (int nt = 0; nt < 2; ++nt)
                acc[mt][nt] = wmma_bf16(aF[mt], bF[nt], acc[mt][nt]);
    }
#pragma unroll
    for (int mt = 0; mt < 4; ++mt)
#pragma unroll
        for (int nt = 0; nt < 2; ++nt)
#pragma unroll
            for (int r = 0; r < 8; ++r) {
                int row = m0 + wm + mt * 16 + r + 8 * hi;
                int col = n0 + wn + nt * 16 + ln;
                float v = acc[mt][nt][r];
                if (EPI == 0)      Cb[(size_t)row * N + col] = (__bf16)v;
                else if (EPI == 1) Cf[(size_t)row * N + col] = v + bias[col] + res[(size_t)row * N + col];
                else               Cf[(size_t)row * N + col] += v + bias[col];
            }
}

// ------- fused GEGLU GEMM: h = (x@W1v+b1v)*gelu(x@W1g+b1g) -------
// Single-buffered LDS: 16 f32 accumulators (128 VGPRs) leave no room for
// long-lived staging regs on gfx1250 (direct encoding caps at VGPR255);
// short store-and-die staging avoids the scratch spills seen with
// double buffering.
__global__ void __launch_bounds__(256)
gemm_geglu(const __bf16* __restrict__ A, const __bf16* __restrict__ B1t,
           const float* __restrict__ b1, __bf16* __restrict__ H) {
    __shared__ __bf16 As [128 * 32];
    __shared__ __bf16 BsV[128 * 32];
    __shared__ __bf16 BsG[128 * 32];
    const int K = DIMN;
    const int tid = threadIdx.x, lane = tid & 31, wid = tid >> 5;
    const int hi = lane >> 4, ln = lane & 15;
    const int wm = (wid >> 2) * 64, wn = (wid & 3) * 32;
    const int m0 = blockIdx.y * 128, n0 = blockIdx.x * 128;

    floatx8 zero8 = {0.f,0.f,0.f,0.f,0.f,0.f,0.f,0.f};
    floatx8 accV[4][2], accG[4][2];
    for (int i = 0; i < 4; ++i) for (int j = 0; j < 2; ++j) { accV[i][j] = zero8; accG[i][j] = zero8; }

    const int r0 = tid >> 2, c8 = (tid & 3) * 8;
    const int r1 = (tid + 256) >> 2;
    const __bf16* Ap0 = A + (size_t)(m0 + r0) * K + c8;
    const __bf16* Ap1 = A + (size_t)(m0 + r1) * K + c8;
    const __bf16* Bv0 = B1t + (size_t)(n0 + r0) * K + c8;
    const __bf16* Bv1 = B1t + (size_t)(n0 + r1) * K + c8;
    const __bf16* Bg0 = B1t + (size_t)(n0 + HIDDEN + r0) * K + c8;
    const __bf16* Bg1 = B1t + (size_t)(n0 + HIDDEN + r1) * K + c8;

    for (int kk = 0; kk < K; kk += 32) {
        // stage tiles (short-lived regs: load -> ds_store -> dead)
        *reinterpret_cast<U4*>(&As [r0 * 32 + c8]) = *reinterpret_cast<const U4*>(Ap0 + kk);
        *reinterpret_cast<U4*>(&As [r1 * 32 + c8]) = *reinterpret_cast<const U4*>(Ap1 + kk);
        *reinterpret_cast<U4*>(&BsV[r0 * 32 + c8]) = *reinterpret_cast<const U4*>(Bv0 + kk);
        *reinterpret_cast<U4*>(&BsV[r1 * 32 + c8]) = *reinterpret_cast<const U4*>(Bv1 + kk);
        *reinterpret_cast<U4*>(&BsG[r0 * 32 + c8]) = *reinterpret_cast<const U4*>(Bg0 + kk);
        *reinterpret_cast<U4*>(&BsG[r1 * 32 + c8]) = *reinterpret_cast<const U4*>(Bg1 + kk);
        __syncthreads();
        Frag bV[2], bG[2];
#pragma unroll
        for (int nt = 0; nt < 2; ++nt) {
            int c = wn + nt * 16 + ln;
            bV[nt].q[0] = *reinterpret_cast<const U4*>(&BsV[c * 32 + 8 * hi]);
            bV[nt].q[1] = *reinterpret_cast<const U4*>(&BsV[c * 32 + 16 + 8 * hi]);
            bG[nt].q[0] = *reinterpret_cast<const U4*>(&BsG[c * 32 + 8 * hi]);
            bG[nt].q[1] = *reinterpret_cast<const U4*>(&BsG[c * 32 + 16 + 8 * hi]);
        }
#pragma unroll
        for (int mt = 0; mt < 4; ++mt) {
            Frag aF;                       // one A-frag live at a time
            int r = wm + mt * 16 + ln;
            aF.q[0] = *reinterpret_cast<const U4*>(&As[r * 32 + 8 * hi]);
            aF.q[1] = *reinterpret_cast<const U4*>(&As[r * 32 + 16 + 8 * hi]);
#pragma unroll
            for (int nt = 0; nt < 2; ++nt) {
                accV[mt][nt] = wmma_bf16(aF, bV[nt], accV[mt][nt]);
                accG[mt][nt] = wmma_bf16(aF, bG[nt], accG[mt][nt]);
            }
        }
        __syncthreads();
    }
#pragma unroll
    for (int mt = 0; mt < 4; ++mt)
#pragma unroll
        for (int nt = 0; nt < 2; ++nt)
#pragma unroll
            for (int r = 0; r < 8; ++r) {
                int row = m0 + wm + mt * 16 + r + 8 * hi;
                int col = n0 + wn + nt * 16 + ln;
                float val  = accV[mt][nt][r] + b1[col];
                float gate = accG[mt][nt][r] + b1[col + HIDDEN];
                float gl   = 0.5f * gate * (1.0f + erff(gate * 0.70710678118654752f));
                H[(size_t)row * HIDDEN + col] = (__bf16)(val * gl);
            }
}

// ---------------- sliding-window causal attention (flash style) ----------------
// grid: (SEQ/WIN, HEADS, BATCH); block: 256 (8 waves). Each wave: 16 q-rows.
__global__ void __launch_bounds__(256)
attn_kernel(const __bf16* __restrict__ Q, const __bf16* __restrict__ Kb,
            const __bf16* __restrict__ Vb, const float* __restrict__ rel_bias,
            __bf16* __restrict__ O) {
    __shared__ __bf16 qs[128 * 64];     // 16 KB
    __shared__ __bf16 ks[128 * 64];     // 16 KB (per pass)
    __shared__ __bf16 vTs[64 * 128];    // 16 KB (per pass, transposed)
    __shared__ __bf16 ps[8][16 * 32];   //  8 KB (per-wave P staging)
    __shared__ float  bias_s[256];

    const int n = blockIdx.x, h = blockIdx.y, b = blockIdx.z;
    const int tid = threadIdx.x, lane = tid & 31, wid = tid >> 5;
    const int hi = lane >> 4, ln = lane & 15;
    const float scale = 0.125f;  // 64^-0.5

    bias_s[tid] = rel_bias[h * 256 + tid];

    const size_t qbase = ((size_t)b * SEQL + (size_t)n * 128) * DIMN + h * 64;
#pragma unroll
    for (int it = 0; it < 2; ++it) {
        int s = tid + it * 256;
        int row = s >> 2, c8 = (s & 3) * 8;
        *reinterpret_cast<U4*>(&qs[row * 64 + c8]) =
            *reinterpret_cast<const U4*>(Q + qbase + (size_t)row * DIMN + c8);
    }

    floatx8 zero8 = {0.f,0.f,0.f,0.f,0.f,0.f,0.f,0.f};
    floatx8 o[4];
    for (int t = 0; t < 4; ++t) o[t] = zero8;
    float m_run[8], l_run[8];
    for (int r = 0; r < 8; ++r) { m_run[r] = -3.0e38f; l_run[r] = 0.f; }

    for (int pp = 0; pp < 2; ++pp) {
        __syncthreads();   // previous pass done reading ks/vTs; qs fill done
        const int sblk = n - 1 + pp;
        const bool vblk = (sblk >= 0);
        const size_t kvbase = ((size_t)b * SEQL + (size_t)(vblk ? sblk : 0) * 128) * DIMN + h * 64;
        U4 zq; zq.x = zq.y = zq.z = zq.w = 0u;
#pragma unroll
        for (int it = 0; it < 2; ++it) {
            int s = tid + it * 256;
            int row = s >> 2, c8 = (s & 3) * 8;
            U4 kd = vblk ? *reinterpret_cast<const U4*>(Kb + kvbase + (size_t)row * DIMN + c8) : zq;
            *reinterpret_cast<U4*>(&ks[row * 64 + c8]) = kd;
            Pack8 vd;
            vd.q = vblk ? *reinterpret_cast<const U4*>(Vb + kvbase + (size_t)row * DIMN + c8) : zq;
#pragma unroll
            for (int e = 0; e < 8; ++e) vTs[(c8 + e) * 128 + row] = vd.h[e];
        }
        __syncthreads();

        for (int jc = 0; jc < 4; ++jc) {
            const int jg0 = pp * 128 + jc * 32;
            Frag qa0, qa1;
            const int ri = wid * 16 + ln;
            qa0.q[0] = *reinterpret_cast<const U4*>(&qs[ri * 64 + 8 * hi]);
            qa0.q[1] = *reinterpret_cast<const U4*>(&qs[ri * 64 + 16 + 8 * hi]);
            qa1.q[0] = *reinterpret_cast<const U4*>(&qs[ri * 64 + 32 + 8 * hi]);
            qa1.q[1] = *reinterpret_cast<const U4*>(&qs[ri * 64 + 48 + 8 * hi]);
            floatx8 st[2];
#pragma unroll
            for (int stt = 0; stt < 2; ++stt) {
                int cj = jc * 32 + stt * 16 + ln;
                Frag kb0, kb1;
                kb0.q[0] = *reinterpret_cast<const U4*>(&ks[cj * 64 + 8 * hi]);
                kb0.q[1] = *reinterpret_cast<const U4*>(&ks[cj * 64 + 16 + 8 * hi]);
                kb1.q[0] = *reinterpret_cast<const U4*>(&ks[cj * 64 + 32 + 8 * hi]);
                kb1.q[1] = *reinterpret_cast<const U4*>(&ks[cj * 64 + 48 + 8 * hi]);
                floatx8 s = zero8;
                s = wmma_bf16(qa0, kb0, s);
                s = wmma_bf16(qa1, kb1, s);
                st[stt] = s;
            }
            float lg[2][8], cm[8];
#pragma unroll
            for (int r = 0; r < 8; ++r) {
                const int irow = wid * 16 + r + 8 * hi;
                float best = -3.0e38f;
#pragma unroll
                for (int stt = 0; stt < 2; ++stt) {
                    int j = jg0 + stt * 16 + ln;
                    int dist = irow + 128 - j;
                    bool valid = (dist >= 0) && ((n > 0) || (j >= 128));
                    int bidx = dist < 0 ? 0 : (dist > 255 ? 255 : dist);
                    float v = valid ? st[stt][r] * scale + bias_s[bidx] : -1.0e9f;
                    lg[stt][r] = v;
                    best = fmaxf(best, v);
                }
                cm[r] = best;
            }
#pragma unroll
            for (int r = 0; r < 8; ++r) {
                float v = cm[r];
                v = fmaxf(v, __shfl_xor(v, 1, 32));
                v = fmaxf(v, __shfl_xor(v, 2, 32));
                v = fmaxf(v, __shfl_xor(v, 4, 32));
                v = fmaxf(v, __shfl_xor(v, 8, 32));
                cm[r] = v;
            }
            float sc[8];
#pragma unroll
            for (int r = 0; r < 8; ++r) {
                float mn = fmaxf(m_run[r], cm[r]);
                sc[r] = __expf(m_run[r] - mn);
                m_run[r] = mn;
            }
            float psum[8];
#pragma unroll
            for (int r = 0; r < 8; ++r) {
                float p0 = __expf(lg[0][r] - m_run[r]);
                float p1 = __expf(lg[1][r] - m_run[r]);
                ps[wid][(r + 8 * hi) * 32 + ln]      = (__bf16)p0;
                ps[wid][(r + 8 * hi) * 32 + 16 + ln] = (__bf16)p1;
                psum[r] = p0 + p1;
            }
#pragma unroll
            for (int r = 0; r < 8; ++r) {
                float v = psum[r];
                v += __shfl_xor(v, 1, 32);
                v += __shfl_xor(v, 2, 32);
                v += __shfl_xor(v, 4, 32);
                v += __shfl_xor(v, 8, 32);
                l_run[r] = l_run[r] * sc[r] + v;
            }
#pragma unroll
            for (int t = 0; t < 4; ++t)
#pragma unroll
                for (int r = 0; r < 8; ++r) o[t][r] *= sc[r];
            // P @ V chunk (same-wave LDS ops are in-order; no barrier needed)
            Frag pA;
            pA.q[0] = *reinterpret_cast<const U4*>(&ps[wid][ln * 32 + 8 * hi]);
            pA.q[1] = *reinterpret_cast<const U4*>(&ps[wid][ln * 32 + 16 + 8 * hi]);
#pragma unroll
            for (int t = 0; t < 4; ++t) {
                int vc = t * 16 + ln;
                Frag vB;
                vB.q[0] = *reinterpret_cast<const U4*>(&vTs[vc * 128 + jc * 32 + 8 * hi]);
                vB.q[1] = *reinterpret_cast<const U4*>(&vTs[vc * 128 + jc * 32 + 16 + 8 * hi]);
                o[t] = wmma_bf16(pA, vB, o[t]);
            }
        }
    }
    const size_t obase = ((size_t)b * SEQL + (size_t)n * 128) * DIMN + h * 64;
#pragma unroll
    for (int t = 0; t < 4; ++t)
#pragma unroll
        for (int r = 0; r < 8; ++r) {
            int row = wid * 16 + r + 8 * hi;
            int col = t * 16 + ln;
            O[obase + (size_t)row * DIMN + col] = (__bf16)(o[t][r] / l_run[r]);
        }
}

// ---------------- host orchestration ----------------
extern "C" void kernel_launch(void* const* d_in, const int* in_sizes, int n_in,
                              void* d_out, int out_size, void* d_ws, size_t ws_size,
                              hipStream_t stream) {
    const float* x        = (const float*)d_in[0];
    const float* ln1_g    = (const float*)d_in[1];
    const float* ln1_b    = (const float*)d_in[2];
    const float* ln2_g    = (const float*)d_in[3];
    const float* ln2_b    = (const float*)d_in[4];
    const float* wq       = (const float*)d_in[5];
    const float* wk       = (const float*)d_in[6];
    const float* wv       = (const float*)d_in[7];
    const float* wo       = (const float*)d_in[8];
    const float* bo       = (const float*)d_in[9];
    const float* rel_bias = (const float*)d_in[10];
    const float* w_ff1    = (const float*)d_in[11];
    const float* b_ff1    = (const float*)d_in[12];
    const float* w_ff2    = (const float*)d_in[13];
    const float* b_ff2    = (const float*)d_in[14];
    float* out = (float*)d_out;
    char*  ws  = (char*)d_ws;

    const size_t MB = 1024ull * 1024ull;
    __bf16* wq_t   = (__bf16*)(ws + 0 * MB);    // 2 MB each, [N,K] layout
    __bf16* wk_t   = (__bf16*)(ws + 2 * MB);
    __bf16* wv_t   = (__bf16*)(ws + 4 * MB);
    __bf16* wo_t   = (__bf16*)(ws + 6 * MB);
    __bf16* wff1_t = (__bf16*)(ws + 8 * MB);    // 8 MB, [4096,1024]
    __bf16* wff2_t = (__bf16*)(ws + 16 * MB);   // 4 MB, [1024,2048]
    __bf16* xn     = (__bf16*)(ws + 20 * MB);   // 32 MB (reused for LN2 output)
    __bf16* qb     = (__bf16*)(ws + 52 * MB);   // 32 MB
    __bf16* kb     = (__bf16*)(ws + 84 * MB);   // 32 MB
    __bf16* vb     = (__bf16*)(ws + 116 * MB);  // 32 MB
    __bf16* attn   = (__bf16*)(ws + 148 * MB);  // 32 MB  (total 180 MB)
    __bf16* hbuf   = (__bf16*)(ws + 52 * MB);   // 64 MB, overlays dead q/k

    // weights -> bf16, transposed to [N,K] so GEMM B-tiles load like A-tiles
    transpose_bf16_kernel<<<dim3(DIMN / 64, DIMN / 64), 256, 0, stream>>>(wq, wq_t, DIMN, DIMN);
    transpose_bf16_kernel<<<dim3(DIMN / 64, DIMN / 64), 256, 0, stream>>>(wk, wk_t, DIMN, DIMN);
    transpose_bf16_kernel<<<dim3(DIMN / 64, DIMN / 64), 256, 0, stream>>>(wv, wv_t, DIMN, DIMN);
    transpose_bf16_kernel<<<dim3(DIMN / 64, DIMN / 64), 256, 0, stream>>>(wo, wo_t, DIMN, DIMN);
    transpose_bf16_kernel<<<dim3(2 * HIDDEN / 64, DIMN / 64), 256, 0, stream>>>(w_ff1, wff1_t, DIMN, 2 * HIDDEN);
    transpose_bf16_kernel<<<dim3(DIMN / 64, HIDDEN / 64), 256, 0, stream>>>(w_ff2, wff2_t, HIDDEN, DIMN);

    // LN1
    layernorm_kernel<<<ROWS, 256, 0, stream>>>(x, ln1_g, ln1_b, xn);

    // QKV projections
    dim3 gqkv(DIMN / 128, ROWS / 128);
    gemm_wmma<0><<<gqkv, 256, 0, stream>>>(xn, wq_t, ROWS, DIMN, DIMN, nullptr, nullptr, qb, nullptr);
    gemm_wmma<0><<<gqkv, 256, 0, stream>>>(xn, wk_t, ROWS, DIMN, DIMN, nullptr, nullptr, kb, nullptr);
    gemm_wmma<0><<<gqkv, 256, 0, stream>>>(xn, wv_t, ROWS, DIMN, DIMN, nullptr, nullptr, vb, nullptr);

    // sliding-window attention
    attn_kernel<<<dim3(SEQL / WINW, HEADS, BATCH), 256, 0, stream>>>(qb, kb, vb, rel_bias, attn);

    // out projection + bias + residual -> d_out (fp32)
    gemm_wmma<1><<<gqkv, 256, 0, stream>>>(attn, wo_t, ROWS, DIMN, DIMN, bo, x, nullptr, out);

    // LN2 on x1 (= d_out)
    layernorm_kernel<<<ROWS, 256, 0, stream>>>(out, ln2_g, ln2_b, xn);

    // fused FF1 + GEGLU -> hbuf (bf16, ROWS x HIDDEN)
    gemm_geglu<<<dim3(HIDDEN / 128, ROWS / 128), 256, 0, stream>>>(xn, wff1_t, b_ff1, hbuf);

    // FF2 + bias, accumulate into d_out
    gemm_wmma<2><<<gqkv, 256, 0, stream>>>(hbuf, wff2_t, ROWS, DIMN, HIDDEN, b_ff2, nullptr, nullptr, out);
}
